// GATDensityRegressor_67731634257977
// MI455X (gfx1250) — matrix operand order repaired and checked by reading, not compiled, not run on hardware
//
#include <hip/hip_runtime.h>
#include <hip/hip_fp16.h>

// ---------------------------------------------------------------------------
// GATv2 (3 layers, 4 heads x 48, concat=False) + linear head + softplus
// MI455X / gfx1250: wave32, WMMA f32_16x16x32_f16 for the node GEMMs.
// All WMMA operands pre-padded f16 so fragment loads are unconditional b128.
// ---------------------------------------------------------------------------

typedef __attribute__((ext_vector_type(16))) _Float16 v16h;
typedef __attribute__((ext_vector_type(8)))  _Float16 v8h;
typedef __attribute__((ext_vector_type(8)))  float    v8f;

#define HID   48
#define HEADS 4
#define DOUT  192   // HEADS*HID
#define DCAT  384   // xl|xr concatenated
#define KP0   32    // padded K for layer 0 (din=3)
#define KP1   64    // padded K for layers 1,2 (din=48)

__device__ __forceinline__ unsigned encodeF(float f) {
    unsigned u = __float_as_uint(f);
    return (u & 0x80000000u) ? ~u : (u | 0x80000000u);
}
__device__ __forceinline__ float decodeF(unsigned u) {
    return (u & 0x80000000u) ? __uint_as_float(u ^ 0x80000000u)
                             : __uint_as_float(~u);
}

// ---- layer-0 activation convert: x[N,3] f32 -> Ah0[N,32] f16 (zero pad) ---
__global__ void cvt_x0(const float* __restrict__ x, _Float16* __restrict__ Ah,
                       int nNodes) {
    int t = blockIdx.x * blockDim.x + threadIdx.x;
    if (t >= nNodes * KP0) return;
    int n = t / KP0, k = t % KP0;
    Ah[t] = (k < 3) ? (_Float16)x[n * 3 + k] : (_Float16)0.0f;
}

// ---- weights: Wl|Wr [din,192]x2 f32 -> WhT[384][kpad] f16 (K-major, pad) --
__global__ void cvt_weightsT(const float* __restrict__ Wl, const float* __restrict__ bl,
                             const float* __restrict__ Wr, const float* __restrict__ br,
                             _Float16* __restrict__ WhT, float* __restrict__ bcat,
                             int din, int kpad) {
    int t = blockIdx.x * blockDim.x + threadIdx.x;
    if (t < DCAT * kpad) {
        int col = t / kpad, k = t % kpad;
        float v = 0.0f;
        if (k < din) v = (col < DOUT) ? Wl[k * DOUT + col] : Wr[k * DOUT + (col - DOUT)];
        WhT[t] = (_Float16)v;
    }
    if (t < DCAT) bcat[t] = (t < DOUT) ? bl[t] : br[t - DOUT];
}

// ---- WMMA GEMM: xlr[N,384] = Ah[N,kpad] * W[kpad,384] + bcat --------------
// 4 waves/block; each wave computes a 16(M) x 128(N) tile, 8 accumulators.
__global__ void __launch_bounds__(128)
gemm_xlxr(const _Float16* __restrict__ Ah, const _Float16* __restrict__ WhT,
          const float* __restrict__ bcat, float* __restrict__ xlr,
          int nNodes, int kpad) {
    const int lane = threadIdx.x & 31;
    const int wv   = threadIdx.x >> 5;
    const int m0   = (blockIdx.x * 4 + wv) * 16;
    if (m0 >= nNodes) return;                 // uniform per wave
    const int n0 = blockIdx.y * 128;

    int row = m0 + (lane & 15);
    if (row >= nNodes) row = nNodes - 1;      // clamp (N % 16 == 0 in practice)

    v8f acc[8];
#pragma unroll
    for (int t = 0; t < 8; ++t) acc[t] = {};

    const _Float16* ap = Ah + (size_t)row * kpad + ((lane >> 4) << 4);

    for (int k0 = 0; k0 < kpad; k0 += 32) {
        // A fragment: two contiguous 8-half runs per lane (together k = 16*(lane/16) + s, same order as B)
        v8h alo = *(const v8h*)(ap + k0);
        v8h ahi = *(const v8h*)(ap + k0 + 8);
        v16h a;
#pragma unroll
        for (int i = 0; i < 8; ++i) { a[i] = alo[i]; a[i + 8] = ahi[i]; }

#pragma unroll
        for (int t = 0; t < 8; ++t) {
            int col = n0 + t * 16 + (lane & 15);
            // B fragment: 16 contiguous halves along K in transposed weights
            v16h b = *(const v16h*)(WhT + (size_t)col * kpad + ((lane >> 4) << 4) + k0);
            acc[t] = __builtin_amdgcn_wmma_f32_16x16x32_f16(
                false, a, false, b, (short)0, acc[t], false, false);
        }
    }

#pragma unroll
    for (int t = 0; t < 8; ++t) {
        int col = n0 + t * 16 + (lane & 15);
        float bv = bcat[col];
#pragma unroll
        for (int j = 0; j < 8; ++j) {
            int r = m0 + j + ((lane >> 4) << 3);
            if (r < nNodes) xlr[(size_t)r * DCAT + col] = acc[t][j] + bv;
        }
    }
}

// ---- per-layer state init -------------------------------------------------
__global__ void init_layer(unsigned* __restrict__ m_enc, float* __restrict__ denom,
                           float* __restrict__ outacc, int nNodes) {
    int t = blockIdx.x * blockDim.x + threadIdx.x;
    if (t < nNodes * DOUT) outacc[t] = 0.0f;
    if (t < nNodes * HEADS) { m_enc[t] = encodeF(-3.0e38f); denom[t] = 0.0f; }
}

// ---- edge logits + running per-(dst,head) max -----------------------------
__global__ void edge_logits(const int* __restrict__ ei, int E, int Ep,
                            const float* __restrict__ xlr, const float* __restrict__ att,
                            float* __restrict__ elog, unsigned* __restrict__ m_enc) {
    int t = blockIdx.x * blockDim.x + threadIdx.x;
    if (t >= Ep * HEADS) return;
    int e = t >> 2, h = t & 3;
    int src, dst;
    if (e < E) { src = ei[e]; dst = ei[E + e]; } else { src = dst = e - E; }
    const float* pl = xlr + (size_t)src * DCAT + h * HID;          // xl[src]
    const float* pr = xlr + (size_t)dst * DCAT + DOUT + h * HID;   // xr[dst]
    const float* pa = att + h * HID;
    float s = 0.0f;
#pragma unroll 4
    for (int c = 0; c < HID; ++c) {
        float v = pl[c] + pr[c];
        v = (v > 0.0f) ? v : 0.2f * v;          // leaky_relu, slope 0.2
        s += pa[c] * v;
    }
    elog[t] = s;
    atomicMax(&m_enc[dst * HEADS + h], encodeF(s));
}

// ---- exp(logit - max) + denom accumulation --------------------------------
__global__ void edge_exp(const int* __restrict__ ei, int E, int Ep,
                         float* __restrict__ elog, const unsigned* __restrict__ m_enc,
                         float* __restrict__ denom) {
    int t = blockIdx.x * blockDim.x + threadIdx.x;
    if (t >= Ep * HEADS) return;
    int e = t >> 2, h = t & 3;
    int dst = (e < E) ? ei[E + e] : (e - E);
    float ex = __expf(elog[t] - decodeF(m_enc[dst * HEADS + h]));
    elog[t] = ex;
    atomicAdd(&denom[dst * HEADS + h], ex);
}

// ---- weighted scatter-add aggregation: one wave per edge ------------------
__global__ void edge_agg(const int* __restrict__ ei, int E, int Ep,
                         const float* __restrict__ xlr, const float* __restrict__ elog,
                         const float* __restrict__ denom, float* __restrict__ outacc) {
    int wid = blockIdx.x * (blockDim.x >> 5) + (threadIdx.x >> 5);
    if (wid >= Ep) return;
    int lane = threadIdx.x & 31;
    int src, dst;
    if (wid < E) { src = ei[wid]; dst = ei[E + wid]; } else { src = dst = wid - E; }
#pragma unroll
    for (int it = 0; it < 6; ++it) {           // 6*32 = 192 channels
        int idx = lane + it * 32;
        int h = idx / HID;
        float alpha = elog[wid * HEADS + h] / (denom[dst * HEADS + h] + 1e-16f);
        atomicAdd(&outacc[(size_t)dst * DOUT + idx],
                  xlr[(size_t)src * DCAT + idx] * alpha);
    }
}

// ---- head-mean + bias + ELU -> xf32[N,48] and padded f16 Ah[N,64] ---------
__global__ void finalize_layer(const float* __restrict__ outacc, const float* __restrict__ bvec,
                               float* __restrict__ xf32, _Float16* __restrict__ Ah,
                               int nNodes) {
    int t = blockIdx.x * blockDim.x + threadIdx.x;
    if (t >= nNodes * KP1) return;
    int n = t / KP1, c = t % KP1;
    if (c < HID) {
        const float* p = outacc + (size_t)n * DOUT + c;
        float s = 0.25f * (p[0] + p[HID] + p[2 * HID] + p[3 * HID]) + bvec[c];
        s = (s > 0.0f) ? s : expm1f(s);        // ELU(alpha=1)
        xf32[(size_t)n * HID + c] = s;
        Ah[t] = (_Float16)s;
    } else {
        Ah[t] = (_Float16)0.0f;                // K padding for next GEMM
    }
}

// ---- linear head + softplus -----------------------------------------------
__global__ void head_kernel(const float* __restrict__ x, const float* __restrict__ hw,
                            const float* __restrict__ hb, float* __restrict__ out, int nNodes) {
    int n = blockIdx.x * blockDim.x + threadIdx.x;
    if (n >= nNodes) return;
    float s = hb[0];
#pragma unroll 4
    for (int c = 0; c < HID; ++c) s += x[(size_t)n * HID + c] * hw[c];
    out[n] = (s > 20.0f) ? s : log1pf(__expf(s));
}

// ---------------------------------------------------------------------------
extern "C" void kernel_launch(void* const* d_in, const int* in_sizes, int n_in,
                              void* d_out, int out_size, void* d_ws, size_t ws_size,
                              hipStream_t stream) {
    const int N  = in_sizes[0] / 3;       // IN_C = 3
    const int E  = in_sizes[1] / 2;
    const int Ep = E + N;                 // edges + self loops

    const float* x_in   = (const float*)d_in[0];
    const int*   ei     = (const int*)d_in[1];
    const float* head_w = (const float*)d_in[20];
    const float* head_b = (const float*)d_in[21];

    // ---- workspace layout -------------------------------------------------
    char* ws = (char*)d_ws;
    size_t off = 0;
    auto carve = [&](size_t bytes) -> void* {
        off = (off + 255) & ~(size_t)255;
        void* p = ws + off;
        off += bytes;
        return p;
    };
    float*     xlr    = (float*)    carve((size_t)N * DCAT * sizeof(float));
    float*     outacc = (float*)    carve((size_t)N * DOUT * sizeof(float));
    float*     xf32   = (float*)    carve((size_t)N * HID  * sizeof(float));
    _Float16*  Ah0    = (_Float16*) carve((size_t)N * KP0  * sizeof(_Float16));
    _Float16*  Ah     = (_Float16*) carve((size_t)N * KP1  * sizeof(_Float16));
    float*     elog   = (float*)    carve((size_t)Ep * HEADS * sizeof(float));
    unsigned*  m_enc  = (unsigned*) carve((size_t)N * HEADS * sizeof(unsigned));
    float*     denom  = (float*)    carve((size_t)N * HEADS * sizeof(float));
    _Float16*  WhT[3];
    float*     bcat[3];
    const int dins [3] = {3, HID, HID};
    const int kpads[3] = {KP0, KP1, KP1};
    for (int l = 0; l < 3; ++l) {
        WhT[l]  = (_Float16*)carve((size_t)DCAT * kpads[l] * sizeof(_Float16));
        bcat[l] = (float*)   carve((size_t)DCAT * sizeof(float));
    }
    (void)ws_size; (void)n_in; (void)out_size;

    const int mtiles = (N + 15) / 16;

    cvt_x0<<<((size_t)N * KP0 + 255) / 256, 256, 0, stream>>>(x_in, Ah0, N);

    for (int l = 0; l < 3; ++l) {
        const int din  = dins[l];
        const int kpad = kpads[l];
        const float* Wl  = (const float*)d_in[2 + 6 * l];
        const float* bl  = (const float*)d_in[3 + 6 * l];
        const float* Wr  = (const float*)d_in[4 + 6 * l];
        const float* br  = (const float*)d_in[5 + 6 * l];
        const float* att = (const float*)d_in[6 + 6 * l];
        const float* bv  = (const float*)d_in[7 + 6 * l];

        cvt_weightsT<<<(DCAT * kpad + 255) / 256, 256, 0, stream>>>(
            Wl, bl, Wr, br, WhT[l], bcat[l], din, kpad);

        const _Float16* Asrc = (l == 0) ? Ah0 : Ah;
        gemm_xlxr<<<dim3((mtiles + 3) / 4, DCAT / 128), 128, 0, stream>>>(
            Asrc, WhT[l], bcat[l], xlr, N, kpad);

        init_layer<<<((size_t)N * DOUT + 255) / 256, 256, 0, stream>>>(
            m_enc, denom, outacc, N);

        int et = Ep * HEADS;
        edge_logits<<<(et + 255) / 256, 256, 0, stream>>>(
            ei, E, Ep, xlr, att, elog, m_enc);

        edge_exp<<<(et + 255) / 256, 256, 0, stream>>>(
            ei, E, Ep, elog, m_enc, denom);

        edge_agg<<<(Ep + 7) / 8, 256, 0, stream>>>(
            ei, E, Ep, xlr, elog, denom, outacc);

        finalize_layer<<<((size_t)N * KP1 + 255) / 256, 256, 0, stream>>>(
            outacc, bv, xf32, Ah, N);
    }

    head_kernel<<<(N + 255) / 256, 256, 0, stream>>>(
        xf32, head_w, head_b, (float*)d_out, N);
}
